// GraphAttention_88270167868035
// MI455X (gfx1250) — matrix operand order repaired and checked by reading, not compiled, not run on hardware
//
#include <hip/hip_runtime.h>
#include <cstdint>

#define NN 8192

typedef __attribute__((ext_vector_type(16))) __bf16 v16bf;
typedef __attribute__((ext_vector_type(8)))  __bf16 v8bf;
typedef __attribute__((ext_vector_type(8)))  float  v8f;
typedef __attribute__((ext_vector_type(4)))  int    v4i;

__device__ __forceinline__ v8f wmma_bf16(v16bf a, v16bf b, v8f c) {
  // D = A(16x32 bf16) x B(32x16 bf16) + C(16x16 f32)
  return __builtin_amdgcn_wmma_f32_16x16x32_bf16(false, a, false, b, (short)0, c,
                                                 false, false);
}

__device__ __forceinline__ v16bf bfcat(v8bf lo, v8bf hi) {
  return __builtin_shufflevector(lo, hi, 0, 1, 2, 3, 4, 5, 6, 7, 8, 9, 10, 11, 12,
                                 13, 14, 15);
}

// CDNA5 async global->LDS copy (ASYNCcnt-tracked), 16B per lane.
__device__ __forceinline__ void async_load_b128_to_lds(void* lds, const void* g) {
  unsigned l = (unsigned)(uintptr_t)lds;                 // low 32 bits = LDS offset
  unsigned long long ga = (unsigned long long)(uintptr_t)g;
  asm volatile("global_load_async_to_lds_b128 %0, %1, off" : : "v"(l), "v"(ga)
               : "memory");
}
__device__ __forceinline__ void wait_asynccnt0() {
  asm volatile("s_wait_asynccnt 0" : : : "memory");
}

__device__ __forceinline__ void atomicMaxF(float* addr, float val) {
  unsigned* ua = (unsigned*)addr;
  unsigned old = __float_as_uint(*addr);
  while (__uint_as_float(old) < val) {
    unsigned assumed = old;
    old = atomicCAS(ua, assumed, __float_as_uint(val));
    if (old == assumed) break;
  }
}

// ---------------- hp init ----------------
__global__ __launch_bounds__(256) void init_hp_kernel(float* hp) {
  hp[threadIdx.x] = -3.0e38f;
}

// ---------------- value1 = x @ W1^T + b1  (K=3, trivial -> VALU) ----------------
__global__ __launch_bounds__(256) void value1_kernel(const float* __restrict__ x,
                                                     const float* __restrict__ W1,
                                                     const float* __restrict__ b1,
                                                     __bf16* __restrict__ V1) {
  int i = blockIdx.x * 256 + threadIdx.x;  // NN*128 total
  int n = i >> 7, d = i & 127;
  float s = b1[d] + x[n * 3 + 0] * W1[d * 3 + 0] + x[n * 3 + 1] * W1[d * 3 + 1] +
            x[n * 3 + 2] * W1[d * 3 + 2];
  V1[i] = (__bf16)s;
}

// ---------------- W2 fp32 -> bf16 ----------------
__global__ __launch_bounds__(256) void w2conv_kernel(const float* __restrict__ W2,
                                                     __bf16* __restrict__ W2b) {
  int i = blockIdx.x * 256 + threadIdx.x;  // 256*128
  W2b[i] = (__bf16)W2[i];
}

// ---------------- attention block 1 ----------------
// scores from x.xT (fp32 VALU, D_IN=3); P@V1 via bf16 WMMA; LN+ReLU -> h1 (bf16)
__global__ __launch_bounds__(128) void attn1_kernel(
    const float* __restrict__ x, const int* __restrict__ adj,
    const __bf16* __restrict__ V1, const float* __restrict__ g1,
    const float* __restrict__ beta1, __bf16* __restrict__ h1) {
  __shared__ __align__(32) __bf16 Vt[128 * 48];  // V tile transposed [col][48]
  __shared__ __align__(16) float xt[96];         // 32 key rows x 3
  __shared__ float rs[4][16];

  const int t = threadIdx.x;
  const int wave = t >> 5;
  const int lane = t & 31;
  const int r = lane & 15;  // A/B fragment row / column index
  const int h = lane >> 4;  // half of wave
  const int rbase = blockIdx.x * 64 + wave * 16;
  const int rowg = rbase + r;

  const float xr0 = x[rowg * 3 + 0];
  const float xr1 = x[rowg * 3 + 1];
  const float xr2 = x[rowg * 3 + 2];
  const float scale = 0.08838834764831845f;  // 1/sqrt(128)

  v8f zero8 = {0.f, 0.f, 0.f, 0.f, 0.f, 0.f, 0.f, 0.f};
  v8f acc[8];
#pragma unroll
  for (int c = 0; c < 8; ++c) acc[c] = zero8;
  float rowsum = 0.0f;

  for (int j0 = 0; j0 < NN; j0 += 32) {
    __syncthreads();
    if (t < 24)  // 96 floats via async global->LDS
      async_load_b128_to_lds(&xt[t * 4], x + (size_t)j0 * 3 + t * 4);
#pragma unroll
    for (int rep = 0; rep < 4; ++rep) {  // stage V tile transposed
      int q = t + rep * 128;
      int row = q >> 4;
      int cc = (q & 15) * 8;
      v8bf v = *(const v8bf*)(V1 + (size_t)(j0 + row) * 128 + cc);
#pragma unroll
      for (int i = 0; i < 8; ++i) Vt[(cc + i) * 48 + row] = v[i];
    }
    if (t < 24) wait_asynccnt0();
    __syncthreads();

    if (j0 + 32 < NN) __builtin_prefetch(V1 + (size_t)(j0 + 32) * 128, 0, 0);

    const int* arow = adj + (size_t)rowg * NN + j0;
    v4i a0 = *(const v4i*)(arow + 8 * h);
    v4i a1 = *(const v4i*)(arow + 8 * h + 4);
    v4i a2 = *(const v4i*)(arow + 16 + 8 * h);
    v4i a3 = *(const v4i*)(arow + 16 + 8 * h + 4);
    int am[16];
#pragma unroll
    for (int i = 0; i < 4; ++i) {
      am[i] = a0[i]; am[4 + i] = a1[i]; am[8 + i] = a2[i]; am[12 + i] = a3[i];
    }

    v16bf p;  // A-fragment: P(16x32) bf16
#pragma unroll
    for (int e = 0; e < 16; ++e) {
      int jloc = (e < 8) ? (e + 8 * h) : (e + 8 + 8 * h);
      float s = (xr0 * xt[jloc * 3 + 0] + xr1 * xt[jloc * 3 + 1] +
                 xr2 * xt[jloc * 3 + 2]) * scale;
      float pe = am[e] ? __expf(s) : 0.0f;
      rowsum += pe;
      p[e] = (__bf16)pe;
    }

#pragma unroll
    for (int c = 0; c < 8; ++c) {  // acc(16x128) += P @ Vtile
      v16bf b = bfcat(*(const v8bf*)(&Vt[(c * 16 + r) * 48 + 8 * h]),
                      *(const v8bf*)(&Vt[(c * 16 + r) * 48 + 16 + 8 * h]));
      acc[c] = wmma_bf16(p, b, acc[c]);
    }
  }

  rowsum += __shfl_xor(rowsum, 16, 32);
  if (h == 0) rs[wave][r] = rowsum;
  __builtin_amdgcn_wave_barrier();

#pragma unroll
  for (int e = 0; e < 8; ++e) {
    float inv = 1.0f / fmaxf(rs[wave][e + 8 * h], 1e-30f);
    float s = 0.f, q = 0.f;
#pragma unroll
    for (int c = 0; c < 8; ++c) {
      float v = acc[c][e] * inv;
      acc[c][e] = v;
      s += v; q += v * v;
    }
#pragma unroll
    for (int m = 1; m <= 8; m <<= 1) {
      s += __shfl_xor(s, m, 32);
      q += __shfl_xor(q, m, 32);
    }
    float mu = s * (1.f / 128.f);
    float var = q * (1.f / 128.f) - mu * mu;
    float rstd = rsqrtf(var + 1e-6f);
    int rowo = rbase + e + 8 * h;
#pragma unroll
    for (int c = 0; c < 8; ++c) {
      int col = c * 16 + r;
      float v = g1[col] * (acc[c][e] - mu) * rstd + beta1[col];
      h1[(size_t)rowo * 128 + col] = (__bf16)fmaxf(v, 0.f);
    }
  }
}

// ---------------- value2 = h1 @ W2^T + b2 via WMMA ----------------
__global__ __launch_bounds__(32) void value2_kernel(const __bf16* __restrict__ h1,
                                                    const __bf16* __restrict__ W2b,
                                                    const float* __restrict__ b2,
                                                    __bf16* __restrict__ V2) {
  int ri = blockIdx.x, ci = blockIdx.y;
  int lane = threadIdx.x;
  int n = lane & 15, h = lane >> 4;
  v8f acc = {0.f, 0.f, 0.f, 0.f, 0.f, 0.f, 0.f, 0.f};
  int row = ri * 16 + n;
#pragma unroll
  for (int dd = 0; dd < 4; ++dd) {
    int d0 = dd * 32;
    v8bf lo = *(const v8bf*)(h1 + (size_t)row * 128 + d0 + 8 * h);
    v8bf hi = *(const v8bf*)(h1 + (size_t)row * 128 + d0 + 16 + 8 * h);
    v16bf A = bfcat(lo, hi);
    v16bf B = bfcat(*(const v8bf*)(W2b + (size_t)(ci * 16 + n) * 128 + d0 + 8 * h),
                    *(const v8bf*)(W2b + (size_t)(ci * 16 + n) * 128 + d0 + 16 + 8 * h));
    acc = wmma_bf16(A, B, acc);
  }
  float bb = b2[ci * 16 + n];
#pragma unroll
  for (int e = 0; e < 8; ++e)
    V2[(size_t)(ri * 16 + e + 8 * h) * 256 + ci * 16 + n] = (__bf16)(acc[e] + bb);
}

// ---------------- attention block 2 + LN + global max pool ----------------
__global__ __launch_bounds__(128) void attn2_kernel(
    const int* __restrict__ adj, const __bf16* __restrict__ h1,
    const __bf16* __restrict__ V2, const float* __restrict__ g2,
    const float* __restrict__ beta2, float* __restrict__ hp) {
  __shared__ __align__(32) __bf16 V2t[256 * 48];  // V tile transposed
  __shared__ float stile[4][16][32];              // per-wave score tile
  __shared__ float rs[4][16];
  __shared__ float hpmax[256];

  const int t = threadIdx.x;
  const int wave = t >> 5;
  const int lane = t & 31;
  const int r = lane & 15;
  const int h = lane >> 4;
  const int rbase = blockIdx.x * 64 + wave * 16;
  const int rowg = rbase + r;

  for (int i = t; i < 256; i += 128) hpmax[i] = -3.0e38f;

  // preload Q fragments (16 rows x 128, bf16) -- 4 K-chunks of 32
  v16bf aQ[4];
#pragma unroll
  for (int dd = 0; dd < 4; ++dd) {
    int d0 = dd * 32;
    v8bf lo = *(const v8bf*)(h1 + (size_t)rowg * 128 + d0 + 8 * h);
    v8bf hi = *(const v8bf*)(h1 + (size_t)rowg * 128 + d0 + 16 + 8 * h);
    aQ[dd] = bfcat(lo, hi);
  }

  v8f zero8 = {0.f, 0.f, 0.f, 0.f, 0.f, 0.f, 0.f, 0.f};
  v8f acc[16];
#pragma unroll
  for (int c = 0; c < 16; ++c) acc[c] = zero8;
  float rowsum = 0.0f;

  for (int j0 = 0; j0 < NN; j0 += 32) {
    __syncthreads();
#pragma unroll
    for (int rep = 0; rep < 8; ++rep) {  // stage V2 tile (32x256) transposed
      int q = t + rep * 128;
      int row = q >> 5;
      int cc = (q & 31) * 8;
      v8bf v = *(const v8bf*)(V2 + (size_t)(j0 + row) * 256 + cc);
#pragma unroll
      for (int i = 0; i < 8; ++i) V2t[(cc + i) * 48 + row] = v[i];
    }
    __syncthreads();

    if (j0 + 32 < NN) __builtin_prefetch(V2 + (size_t)(j0 + 32) * 256, 0, 0);

    // scores S(16x32) = Q @ K^T via WMMA (two 16x16 tiles)
#pragma unroll
    for (int jt = 0; jt < 2; ++jt) {
      v8f sc = zero8;
#pragma unroll
      for (int dd = 0; dd < 4; ++dd) {
        v16bf B = bfcat(*(const v8bf*)(h1 + (size_t)(j0 + jt * 16 + r) * 128 + dd * 32 +
                                       8 * h),
                        *(const v8bf*)(h1 + (size_t)(j0 + jt * 16 + r) * 128 + dd * 32 +
                                       16 + 8 * h));
        sc = wmma_bf16(aQ[dd], B, sc);
      }
#pragma unroll
      for (int e = 0; e < 8; ++e) stile[wave][e + 8 * h][jt * 16 + r] = sc[e];
    }
    __builtin_amdgcn_wave_barrier();

    const int* arow = adj + (size_t)rowg * NN + j0;
    v4i a0 = *(const v4i*)(arow + 8 * h);
    v4i a1 = *(const v4i*)(arow + 8 * h + 4);
    v4i a2 = *(const v4i*)(arow + 16 + 8 * h);
    v4i a3 = *(const v4i*)(arow + 16 + 8 * h + 4);
    int am[16];
#pragma unroll
    for (int i = 0; i < 4; ++i) {
      am[i] = a0[i]; am[4 + i] = a1[i]; am[8 + i] = a2[i]; am[12 + i] = a3[i];
    }

    v16bf p;
#pragma unroll
    for (int e = 0; e < 16; ++e) {
      int jloc = (e < 8) ? (e + 8 * h) : (e + 8 + 8 * h);
      float s = stile[wave][r][jloc] * 0.0625f;  // 1/sqrt(256)
      float pe = am[e] ? __expf(s) : 0.0f;
      rowsum += pe;
      p[e] = (__bf16)pe;
    }
    __builtin_amdgcn_wave_barrier();

#pragma unroll
    for (int c = 0; c < 16; ++c) {  // acc(16x256) += P @ V2tile
      v16bf b = bfcat(*(const v8bf*)(&V2t[(c * 16 + r) * 48 + 8 * h]),
                      *(const v8bf*)(&V2t[(c * 16 + r) * 48 + 16 + 8 * h]));
      acc[c] = wmma_bf16(p, b, acc[c]);
    }
  }

  rowsum += __shfl_xor(rowsum, 16, 32);
  if (h == 0) rs[wave][r] = rowsum;
  __builtin_amdgcn_wave_barrier();

#pragma unroll
  for (int e = 0; e < 8; ++e) {
    float inv = 1.0f / fmaxf(rs[wave][e + 8 * h], 1e-30f);
    float s = 0.f, q = 0.f;
#pragma unroll
    for (int c = 0; c < 16; ++c) {
      float v = acc[c][e] * inv;
      acc[c][e] = v;
      s += v; q += v * v;
    }
#pragma unroll
    for (int m = 1; m <= 8; m <<= 1) {
      s += __shfl_xor(s, m, 32);
      q += __shfl_xor(q, m, 32);
    }
    float mu = s * (1.f / 256.f);
    float var = q * (1.f / 256.f) - mu * mu;
    float rstd = rsqrtf(var + 1e-6f);
#pragma unroll
    for (int c = 0; c < 16; ++c) {
      int col = c * 16 + r;
      float v = g2[col] * (acc[c][e] - mu) * rstd + beta2[col];
      acc[c][e] = v;
    }
  }
  // per-lane column max over the 8 rows it holds, then block/global reduce
#pragma unroll
  for (int c = 0; c < 16; ++c) {
    float m = acc[c][0];
#pragma unroll
    for (int e = 1; e < 8; ++e) m = fmaxf(m, acc[c][e]);
    atomicMaxF(&hpmax[c * 16 + r], m);
  }
  __syncthreads();
  for (int i = t; i < 256; i += 128) atomicMaxF(&hp[i], hpmax[i]);
}

// ---------------- head: f1 = relu(hp@Wf1^T+bf1); out = softmax(f1@Wf2^T+bf2) ----
__global__ __launch_bounds__(384) void head_kernel(const float* __restrict__ hp,
                                                   const float* __restrict__ Wf1,
                                                   const float* __restrict__ bf1v,
                                                   const float* __restrict__ Wf2,
                                                   const float* __restrict__ bf2v,
                                                   float* __restrict__ out) {
  __shared__ float f1[384];
  int t = threadIdx.x;
  float s = bf1v[t];
  for (int k = 0; k < 256; ++k) s += hp[k] * Wf1[t * 256 + k];
  f1[t] = fmaxf(s, 0.f);
  __syncthreads();
  if (t == 0) {
    float o[3], mx = -3.0e38f;
    for (int i = 0; i < 3; ++i) {
      float a = bf2v[i];
      for (int k = 0; k < 384; ++k) a += f1[k] * Wf2[i * 384 + k];
      o[i] = a;
      mx = fmaxf(mx, a);
    }
    float den = 0.f;
    for (int i = 0; i < 3; ++i) den += __expf(o[i] - mx);
    for (int i = 0; i < 3; ++i) out[i] = __expf(o[i] - mx) / den;
  }
}

extern "C" void kernel_launch(void* const* d_in, const int* in_sizes, int n_in,
                              void* d_out, int out_size, void* d_ws, size_t ws_size,
                              hipStream_t stream) {
  (void)in_sizes; (void)n_in; (void)out_size; (void)ws_size;
  const float* x     = (const float*)d_in[0];
  const int*   adj   = (const int*)d_in[1];
  const float* W1    = (const float*)d_in[2];
  const float* b1    = (const float*)d_in[3];
  const float* g1    = (const float*)d_in[4];
  const float* beta1 = (const float*)d_in[5];
  const float* W2    = (const float*)d_in[6];
  const float* b2    = (const float*)d_in[7];
  const float* g2    = (const float*)d_in[8];
  const float* beta2 = (const float*)d_in[9];
  const float* Wf1   = (const float*)d_in[10];
  const float* bf1v  = (const float*)d_in[11];
  const float* Wf2   = (const float*)d_in[12];
  const float* bf2v  = (const float*)d_in[13];

  char* ws = (char*)d_ws;
  __bf16* V1  = (__bf16*)(ws);                 // 8192*128*2 = 2 MiB
  __bf16* h1  = (__bf16*)(ws + (2u << 20));    // 2 MiB
  __bf16* V2  = (__bf16*)(ws + (4u << 20));    // 4 MiB
  __bf16* W2b = (__bf16*)(ws + (8u << 20));    // 64 KiB
  float*  hp  = (float*)(ws + (8u << 20) + 65536);

  init_hp_kernel<<<1, 256, 0, stream>>>(hp);
  value1_kernel<<<(NN * 128) / 256, 256, 0, stream>>>(x, W1, b1, V1);
  w2conv_kernel<<<(256 * 128) / 256, 256, 0, stream>>>(W2, W2b);
  attn1_kernel<<<NN / 64, 128, 0, stream>>>(x, adj, V1, g1, beta1, h1);
  value2_kernel<<<dim3(NN / 16, 16), 32, 0, stream>>>(h1, W2b, b2, V2);
  attn2_kernel<<<NN / 64, 128, 0, stream>>>(adj, h1, V2, g2, beta2, hp);
  head_kernel<<<1, 384, 0, stream>>>(hp, Wf1, bf1v, Wf2, bf2v, (float*)d_out);
}
